// PolicyGGNN_111669150310
// MI455X (gfx1250) — compile-verified, matrix operand order checked by reading, not run on hardware
//
#include <hip/hip_runtime.h>
#include <hip/hip_bf16.h>
#include <math.h>

// Problem constants (from reference)
#define NN    10000   // nodes
#define EE    160000  // edges
#define OUT_C 1000    // channels
#define NB    16      // graphs in batch
#define MP    10112   // padded rows = 79 * 128
#define KP    1024    // padded K
#define N3    3072    // padded 3*OUT

#define LDS_ROW   40                  // 32 bf16 + 8 pad (16B-aligned chunks)
#define LDS_TILE  (128 * LDS_ROW)     // elements per tile buffer
#define LDS_TILE_B (LDS_TILE * 2)     // bytes per tile buffer

typedef __attribute__((ext_vector_type(16))) __bf16 v16bf;
typedef __attribute__((ext_vector_type(8)))  float  v8f;

__device__ __forceinline__ unsigned short f2bf(float f) {
  unsigned u = __float_as_uint(f);
  u += 0x7FFFu + ((u >> 16) & 1u);   // round-to-nearest-even
  return (unsigned short)(u >> 16);
}

// order-preserving float<->uint map for atomicMax-based float max
__device__ __forceinline__ unsigned f2ord(float f) {
  unsigned u = __float_as_uint(f);
  return (u & 0x80000000u) ? ~u : (u | 0x80000000u);
}
__device__ __forceinline__ float ord2f(unsigned u) {
  unsigned b = (u & 0x80000000u) ? (u ^ 0x80000000u) : ~u;
  return __uint_as_float(b);
}

// CDNA5 async DMA: global -> LDS, 16B per lane, tracked by ASYNCcnt.
// vdst = per-lane LDS byte address, vaddr = 64-bit global address, no SADDR.
__device__ __forceinline__ void async_ld_b128(unsigned lds_off, const void* gptr) {
  asm volatile("global_load_async_to_lds_b128 %0, %1, off"
               :: "v"(lds_off), "v"((unsigned long long)(size_t)gptr)
               : "memory");
}
__device__ __forceinline__ void wait_async0() {
#if __has_builtin(__builtin_amdgcn_s_wait_asynccnt)
  __builtin_amdgcn_s_wait_asynccnt(0);
#else
  asm volatile("s_wait_asynccnt 0" ::: "memory");
#endif
}

// ---------------------------------------------------------------------------
// h = pad(x, [N, OUT]) ; also bf16 padded copy (h, Hb pre-zeroed by memset)
// ---------------------------------------------------------------------------
__global__ void init_h_kernel(const float* __restrict__ x,
                              float* __restrict__ h,
                              unsigned short* __restrict__ Hb) {
  int i = blockIdx.x * blockDim.x + threadIdx.x;
  if (i >= NN * 4) return;
  int node = i >> 2, c = i & 3;
  float v = x[i];
  h[(size_t)node * OUT_C + c] = v;
  Hb[(size_t)node * KP + c]   = f2bf(v);
}

// BW[l][n][k] = W[l][k][n], zero-padded, bf16 (K-major rows for WMMA B tiles)
__global__ void prep_W_kernel(const float* __restrict__ W,
                              unsigned short* __restrict__ BW) {
  int i = blockIdx.x * blockDim.x + threadIdx.x;
  if (i >= 3 * KP * KP) return;
  int l = i / (KP * KP);
  int rem = i - l * KP * KP;
  int n = rem / KP;
  int k = rem - n * KP;
  float v = 0.f;
  if (n < OUT_C && k < OUT_C) v = W[((size_t)l * OUT_C + k) * OUT_C + n];
  BW[i] = f2bf(v);
}

// Bp[r][k] = Wsrc[r][k] zero-padded to [3072][1024] bf16 (w_ih / w_hh)
__global__ void prep_pad_kernel(const float* __restrict__ Wsrc,
                                unsigned short* __restrict__ Bp, int rows) {
  int i = blockIdx.x * blockDim.x + threadIdx.x;
  if (i >= N3 * KP) return;
  int r = i / KP, k = i - r * KP;
  float v = 0.f;
  if (r < rows && k < OUT_C) v = Wsrc[(size_t)r * OUT_C + k];
  Bp[i] = f2bf(v);
}

// fp32 [N x OUT] -> bf16 padded [MP x KP] (full write, zeros in pad)
__global__ void cvt_bf16_pad_kernel(const float* __restrict__ src,
                                    unsigned short* __restrict__ dstb) {
  long long i = (long long)blockIdx.x * blockDim.x + threadIdx.x;
  if (i >= (long long)MP * KP) return;
  int r = (int)(i / KP);
  int c = (int)(i - (long long)r * KP);
  float v = 0.f;
  if (r < NN && c < OUT_C) v = src[(size_t)r * OUT_C + c];
  dstb[i] = f2bf(v);
}

// ---------------------------------------------------------------------------
// bf16 WMMA GEMM: C[M x Ncols] = A[MP x KP] * B^T, B stored N-major [Nn x KP].
// Block tile 128x128, 8 waves, wave tile 32x64 (2x4 16x16 frags), K step 32.
// Async global->LDS staging (ASYNCcnt) with LDS double buffering:
// one barrier per K step; top-of-iter (wait_asynccnt + barrier) both publishes
// the prefetched tile and proves every wave finished reading the other buffer.
// ---------------------------------------------------------------------------
__global__ __launch_bounds__(256) void gemm_bf16_wmma(
    const unsigned short* __restrict__ A,
    const unsigned short* __restrict__ B,
    float* __restrict__ C,
    const float* __restrict__ bias,   // [Ncols] or nullptr
    int Mreal, int Ncols) {
  __shared__ __align__(16) unsigned short As[2][LDS_TILE];
  __shared__ __align__(16) unsigned short Bs[2][LDS_TILE];

  const int tileM = blockIdx.x * 128;
  const int tileN = blockIdx.y * 128;
  const int tid   = threadIdx.x;
  const int wave  = tid >> 5;
  const int lane  = tid & 31;
  const int wm    = wave >> 1;   // 0..3 : M quadrant
  const int wn    = wave & 1;    // 0..1 : N half
  const int lm    = lane & 15;
  const int kh    = lane >> 4;   // K-half select per ISA 16-bit A layout

  // staging: thread owns elements {tid, tid+256} of the 512 16B-chunks per tile
  const int i0 = tid, i1 = tid + 256;
  const int r0 = i0 >> 2, c0 = (i0 & 3) * 8;
  const int r1 = i1 >> 2, c1 = (i1 & 3) * 8;
  const unsigned short* ga0 = A + (size_t)(tileM + r0) * KP + c0;
  const unsigned short* ga1 = A + (size_t)(tileM + r1) * KP + c1;
  const unsigned short* gb0 = B + (size_t)(tileN + r0) * KP + c0;
  const unsigned short* gb1 = B + (size_t)(tileN + r1) * KP + c1;
  const unsigned la0 = (unsigned)(size_t)&As[0][r0 * LDS_ROW + c0];
  const unsigned la1 = (unsigned)(size_t)&As[0][r1 * LDS_ROW + c1];
  const unsigned lb0 = (unsigned)(size_t)&Bs[0][r0 * LDS_ROW + c0];
  const unsigned lb1 = (unsigned)(size_t)&Bs[0][r1 * LDS_ROW + c1];

  v8f acc[2][4];
#pragma unroll
  for (int a = 0; a < 2; ++a)
#pragma unroll
    for (int b = 0; b < 4; ++b)
#pragma unroll
      for (int r = 0; r < 8; ++r) acc[a][b][r] = 0.0f;

  // prologue: prefetch K-step 0 into buffer 0
  async_ld_b128(la0, ga0);
  async_ld_b128(la1, ga1);
  async_ld_b128(lb0, gb0);
  async_ld_b128(lb1, gb1);

  int buf = 0;
  for (int k0 = 0; k0 < KP; k0 += 32) {
    wait_async0();
    __syncthreads();

    if (k0 + 32 < KP) {           // prefetch next K step into the other buffer
      const unsigned bo = (unsigned)((buf ^ 1) * LDS_TILE_B);
      const size_t go = (size_t)(k0 + 32);
      async_ld_b128(la0 + bo, ga0 + go);
      async_ld_b128(la1 + bo, ga1 + go);
      async_ld_b128(lb0 + bo, gb0 + go);
      async_ld_b128(lb1 + bo, gb1 + go);
    }

    // fragment loads: lanes 0-15 take K {0..7,16..23}, lanes 16-31 {8..15,24..31}
    const unsigned short* as = As[buf];
    const unsigned short* bs = Bs[buf];
    v16bf af[2], bfr[4];
#pragma unroll
    for (int fm = 0; fm < 2; ++fm) {
      const int row = wm * 32 + fm * 16 + lm;
      union { uint4 u[2]; v16bf v; } t;
      t.u[0] = *(const uint4*)(&as[row * LDS_ROW + kh * 8]);
      t.u[1] = *(const uint4*)(&as[row * LDS_ROW + 16 + kh * 8]);
      af[fm] = t.v;
    }
#pragma unroll
    for (int fn = 0; fn < 4; ++fn) {
      const int row = wn * 64 + fn * 16 + lm;
      union { uint4 u[2]; v16bf v; } t;
      t.u[0] = *(const uint4*)(&bs[row * LDS_ROW + kh * 8]);
      t.u[1] = *(const uint4*)(&bs[row * LDS_ROW + 16 + kh * 8]);
      bfr[fn] = t.v;
    }
#pragma unroll
    for (int fm = 0; fm < 2; ++fm)
#pragma unroll
      for (int fn = 0; fn < 4; ++fn)
        acc[fm][fn] = __builtin_amdgcn_wmma_f32_16x16x32_bf16(
            false, af[fm], false, bfr[fn], (short)0, acc[fm][fn], false, false);

    buf ^= 1;
  }

  // epilogue: lanes 0-15 -> M=r, N=lane ; lanes 16-31 -> M=r+8, N=lane-16
#pragma unroll
  for (int fm = 0; fm < 2; ++fm) {
#pragma unroll
    for (int fn = 0; fn < 4; ++fn) {
      const int gc = tileN + wn * 64 + fn * 16 + lm;
#pragma unroll
      for (int r = 0; r < 8; ++r) {
        const int gr = tileM + wm * 32 + fm * 16 + kh * 8 + r;
        if (gr < Mreal && gc < Ncols) {
          float v = acc[fm][fn][r];
          if (bias) v += bias[gc];
          C[(size_t)gr * Ncols + gc] = v;
        }
      }
    }
  }
}

// ---------------------------------------------------------------------------
// agg[dst] += m[src] * ew  (fp32 atomics, float4 per thread)
// ---------------------------------------------------------------------------
__global__ void scatter_kernel(const float* __restrict__ m,
                               const int* __restrict__ src,
                               const int* __restrict__ dst,
                               const float* __restrict__ ew,
                               float* __restrict__ agg) {
  int tid = blockIdx.x * blockDim.x + threadIdx.x;
  if (tid >= EE * (OUT_C / 4)) return;
  int e = tid / (OUT_C / 4);
  int c = (tid - e * (OUT_C / 4)) * 4;
  float w = ew[e];
  int s = src[e], d = dst[e];
  float4 mv = *(const float4*)(m + (size_t)s * OUT_C + c);
  float* ap = agg + (size_t)d * OUT_C + c;
  atomicAdd(ap + 0, mv.x * w);
  atomicAdd(ap + 1, mv.y * w);
  atomicAdd(ap + 2, mv.z * w);
  atomicAdd(ap + 3, mv.w * w);
}

// ---------------------------------------------------------------------------
// GRU cell elementwise; gi/gh already contain biases from GEMM epilogue
// ---------------------------------------------------------------------------
__global__ void gru_kernel(const float* __restrict__ gi,
                           const float* __restrict__ gh,
                           float* __restrict__ h,
                           unsigned short* __restrict__ Hb) {
  int i = blockIdx.x * blockDim.x + threadIdx.x;
  if (i >= NN * OUT_C) return;
  int node = i / OUT_C, c = i - node * OUT_C;
  size_t b = (size_t)node * (3 * OUT_C) + c;
  float ir = gi[b], iz = gi[b + OUT_C], in_ = gi[b + 2 * OUT_C];
  float hr = gh[b], hz = gh[b + OUT_C], hn  = gh[b + 2 * OUT_C];
  float r = 1.f / (1.f + __expf(-(ir + hr)));
  float z = 1.f / (1.f + __expf(-(iz + hz)));
  float n = tanhf(in_ + r * hn);
  float hv = (1.f - z) * n + z * h[i];
  h[i] = hv;
  Hb[(size_t)node * KP + c] = f2bf(hv);
}

// ---------------------------------------------------------------------------
// logits[i] = relu(h[i]) . fc_w + fc_b  (one wave32 per node)
// ---------------------------------------------------------------------------
__global__ __launch_bounds__(128) void logits_kernel(
    const float* __restrict__ h, const float* __restrict__ fcw,
    const float* __restrict__ fcb, float* __restrict__ logits) {
  int node = blockIdx.x * 4 + (threadIdx.x >> 5);
  int lane = threadIdx.x & 31;
  if (node >= NN) return;
  float s = 0.f;
  for (int c = lane; c < OUT_C; c += 32)
    s += fmaxf(h[(size_t)node * OUT_C + c], 0.f) * fcw[c];
#pragma unroll
  for (int off = 16; off > 0; off >>= 1) s += __shfl_down(s, off, 32);
  if (lane == 0) logits[node] = s + fcb[0];
}

// ---------------------------------------------------------------------------
// masked segment softmax over batch ids (B=16), single workgroup
// ---------------------------------------------------------------------------
__global__ __launch_bounds__(256) void softmax_kernel(
    const float* __restrict__ logits, const int* __restrict__ mask,
    const int* __restrict__ batch, float* __restrict__ out) {
  __shared__ unsigned smax[NB];
  __shared__ float ssum[NB];
  __shared__ float segm[NB];
  int tid = threadIdx.x;
  if (tid < NB) { smax[tid] = 0u; ssum[tid] = 0.f; }
  __syncthreads();

  unsigned lmax[NB];
  for (int b = 0; b < NB; ++b) lmax[b] = 0u;
  for (int i = tid; i < NN; i += 256) {
    if (mask[i]) {
      int b = batch[i];
      unsigned o = f2ord(logits[i]);
      if (o > lmax[b]) lmax[b] = o;
    }
  }
  for (int b = 0; b < NB; ++b)
    if (lmax[b]) atomicMax(&smax[b], lmax[b]);
  __syncthreads();

  if (tid < NB) segm[tid] = (smax[tid] == 0u) ? 0.f : ord2f(smax[tid]);
  __syncthreads();

  float lsum[NB];
  for (int b = 0; b < NB; ++b) lsum[b] = 0.f;
  for (int i = tid; i < NN; i += 256) {
    if (mask[i]) {
      int b = batch[i];
      lsum[b] += __expf(logits[i] - segm[b]);
    }
  }
  for (int b = 0; b < NB; ++b)
    if (lsum[b] != 0.f) atomicAdd(&ssum[b], lsum[b]);
  __syncthreads();

  for (int i = tid; i < NN; i += 256) {
    int b = batch[i];
    float e = mask[i] ? __expf(logits[i] - segm[b]) : 0.f;
    out[i] = e / (ssum[b] + 1e-16f);
  }
}

// ---------------------------------------------------------------------------
extern "C" void kernel_launch(void* const* d_in, const int* in_sizes, int n_in,
                              void* d_out, int out_size, void* d_ws, size_t ws_size,
                              hipStream_t stream) {
  const float* x    = (const float*)d_in[0];
  const int*   ei   = (const int*)d_in[1];
  const float* ew   = (const float*)d_in[2];
  const int*   mask = (const int*)d_in[3];
  const int*   batc = (const int*)d_in[4];
  const float* W    = (const float*)d_in[5];
  const float* wih  = (const float*)d_in[6];
  const float* whh  = (const float*)d_in[7];
  const float* bih  = (const float*)d_in[8];
  const float* bhh  = (const float*)d_in[9];
  const float* fcw  = (const float*)d_in[10];
  const float* fcb  = (const float*)d_in[11];
  float* out = (float*)d_out;

  char* ws = (char*)d_ws;
  size_t off = 0;
  auto alloc = [&](size_t bytes) -> char* {
    char* p = ws + off;
    off += (bytes + 255) & ~(size_t)255;
    return p;
  };
  float*          h      = (float*)alloc((size_t)NN * OUT_C * 4);
  unsigned short* Hb     = (unsigned short*)alloc((size_t)MP * KP * 2);
  unsigned short* Ab     = (unsigned short*)alloc((size_t)MP * KP * 2);
  float*          gi     = (float*)alloc((size_t)NN * 3 * OUT_C * 4);
  float*          gh     = (float*)alloc((size_t)NN * 3 * OUT_C * 4);
  unsigned short* BW     = (unsigned short*)alloc((size_t)3 * KP * KP * 2);
  unsigned short* Bih    = (unsigned short*)alloc((size_t)N3 * KP * 2);
  unsigned short* Bhh    = (unsigned short*)alloc((size_t)N3 * KP * 2);
  float*          logits = (float*)alloc((size_t)NN * 4);
  // lifetime-safe aliases (m dead after scatter; agg dead after cvt)
  float* m   = gi;
  float* agg = gh;

  const int* src = ei;
  const int* dst = ei + EE;

  hipMemsetAsync(h, 0, (size_t)NN * OUT_C * 4, stream);
  hipMemsetAsync(Hb, 0, (size_t)MP * KP * 2, stream);

  init_h_kernel<<<(NN * 4 + 255) / 256, 256, 0, stream>>>(x, h, Hb);
  prep_W_kernel<<<(3 * KP * KP + 255) / 256, 256, 0, stream>>>(W, BW);
  prep_pad_kernel<<<(N3 * KP + 255) / 256, 256, 0, stream>>>(wih, Bih, 3 * OUT_C);
  prep_pad_kernel<<<(N3 * KP + 255) / 256, 256, 0, stream>>>(whh, Bhh, 3 * OUT_C);

  for (int l = 0; l < 3; ++l) {
    // m = h @ W[l]
    gemm_bf16_wmma<<<dim3(MP / 128, KP / 128), 256, 0, stream>>>(
        Hb, BW + (size_t)l * KP * KP, m, nullptr, NN, OUT_C);
    // agg = segment_sum(m[src] * ew, dst)
    hipMemsetAsync(agg, 0, (size_t)NN * OUT_C * 4, stream);
    scatter_kernel<<<(EE * (OUT_C / 4) + 255) / 256, 256, 0, stream>>>(
        m, src, dst, ew, agg);
    cvt_bf16_pad_kernel<<<(unsigned)(((size_t)MP * KP + 255) / 256), 256, 0, stream>>>(agg, Ab);
    // gi = agg @ w_ih^T + b_ih ; gh = h @ w_hh^T + b_hh
    gemm_bf16_wmma<<<dim3(MP / 128, N3 / 128), 256, 0, stream>>>(
        Ab, Bih, gi, bih, NN, 3 * OUT_C);
    gemm_bf16_wmma<<<dim3(MP / 128, N3 / 128), 256, 0, stream>>>(
        Hb, Bhh, gh, bhh, NN, 3 * OUT_C);
    gru_kernel<<<(NN * OUT_C + 255) / 256, 256, 0, stream>>>(gi, gh, h, Hb);
  }

  logits_kernel<<<(NN + 3) / 4, 128, 0, stream>>>(h, fcw, fcb, logits);
  softmax_kernel<<<1, 256, 0, stream>>>(logits, mask, batc, out);
}
